// TransformerBlock_64218351009957
// MI455X (gfx1250) — matrix operand order repaired and checked by reading, not compiled, not run on hardware
//
#include <hip/hip_runtime.h>
#include <cmath>

typedef _Float16 f16;
typedef __attribute__((ext_vector_type(16))) _Float16 v16h;
typedef __attribute__((ext_vector_type(8)))  _Float16 v8h;
typedef __attribute__((ext_vector_type(8)))  float    v8f;
typedef __attribute__((ext_vector_type(4)))  unsigned int u32x4;
typedef __attribute__((ext_vector_type(8)))  int i32x8;
typedef __attribute__((ext_vector_type(4)))  int i32x4;

// ---------- helpers ----------
__device__ __forceinline__ float clip1e4(float v) {
  return fminf(fmaxf(v, -10000.f), 10000.f);
}
__device__ __forceinline__ float fin0(float v) {
  return (v == v && fabsf(v) <= 3.0e38f) ? v : 0.f;
}
__device__ __forceinline__ float waveReduceAdd(float v) {
#pragma unroll
  for (int m = 16; m > 0; m >>= 1) v += __shfl_xor(v, m, 32);
  return v;
}
// reduce within each 16-lane half (rows of a WMMA C fragment live across 16 lanes)
__device__ __forceinline__ float halfReduceMax(float v) {
#pragma unroll
  for (int m = 8; m > 0; m >>= 1) v = fmaxf(v, __shfl_xor(v, m, 32));
  return v;
}
__device__ __forceinline__ float halfReduceAdd(float v) {
#pragma unroll
  for (int m = 8; m > 0; m >>= 1) v += __shfl_xor(v, m, 32);
  return v;
}

// ---------- Tensor Data Mover: 2D f16 tile (rows x 32 elems) -> LDS ----------
// D# built per CDNA5 ISA ch.8: group0 = {flags, lds_addr, global_addr, type=2},
// group1 = {data_size=2B + LDS pad (16B per 64B row), tensor/tile dims, dim0 stride}.
// Pad makes LDS row stride 80B -> 16 consecutive rows land in 16 distinct banks.
// Toolchain here is the clang-23 amdgpu-toolchain: 6-arg builtin
// (u32x4 g0, i32x8 g1, i32x4 g2, i32x4 g3, i32x8 extra, i32 cpol).
#define BT_ROW 40  // f16 elements per padded LDS row (32 data + 8 pad)

__device__ __forceinline__ void tdm_load_tile_f16(unsigned lds_off, const void* gsrc,
                                                  int rows, int stride_elems) {
  unsigned long long ga = (unsigned long long)(size_t)gsrc;
  u32x4 g0;
  g0[0] = 1u;                                            // count=1, user-mode load
  g0[1] = lds_off;                                       // lds_addr (bytes)
  g0[2] = (unsigned)ga;                                  // global_addr[31:0]
  g0[3] = (unsigned)((ga >> 32) & 0x01ffffffu) | (2u << 30);  // addr[56:32], type=2
  i32x8 g1;
  g1[0] = (int)((1u << 16) | (1u << 20) | (3u << 22) | (3u << 25));
  //        data_size=2B | pad_enable | pad_interval=16 DW (64B) | pad_amount=4 DW (16B)
  g1[1] = (int)(32u << 16);                              // tensor_dim0 = 32 (tile-exact)
  g1[2] = (int)((unsigned)rows << 16);                   // tensor_dim1 = rows
  g1[3] = (int)(32u << 16);                              // tile_dim0 = 32
  g1[4] = (int)((unsigned)rows & 0xffffu);               // tile_dim1 = rows, tile_dim2 = 0
  g1[5] = (int)stride_elems;                             // tensor_dim0_stride (elems)
  g1[6] = 0;
  g1[7] = 0;
  i32x4 z4 = {0, 0, 0, 0};                               // 2D tensor: groups 2/3 unused
  i32x8 z8 = {0, 0, 0, 0, 0, 0, 0, 0};
  __builtin_amdgcn_tensor_load_to_lds(g0, g1, z4, z4, z8, 0);
}

// ---------- f32 -> f16 weight conversion with zero padding ----------
__global__ __launch_bounds__(256) void cvt_pad_f16_kernel(
    const float* __restrict__ src, f16* __restrict__ dst,
    int srows, int scols, int drows, int dcols) {
  const size_t n = (size_t)drows * dcols;
  const size_t stride = (size_t)gridDim.x * blockDim.x;
  for (size_t i = (size_t)blockIdx.x * blockDim.x + threadIdx.x; i < n; i += stride) {
    const int r = (int)(i / dcols);
    const int c = (int)(i % dcols);
    float v = (r < srows && c < scols) ? src[(size_t)r * scols + c] : 0.f;
    dst[i] = (f16)v;
  }
}

// ---------- rmsnorm over rows of length D, output f16 ----------
__global__ __launch_bounds__(256) void rmsnorm_row_f16_kernel(
    const float* __restrict__ in, const float* __restrict__ w,
    f16* __restrict__ out, int D) {
  __shared__ float red[8];
  __shared__ float sinv;
  const int tid = threadIdx.x;
  const float* p = in + (size_t)blockIdx.x * D;
  float ss = 0.f;
  for (int i = tid; i < D; i += 256) { float v = clip1e4(p[i]); ss += v * v; }
  ss = waveReduceAdd(ss);
  if ((tid & 31) == 0) red[tid >> 5] = ss;
  __syncthreads();
  if (tid == 0) {
    float s = 0.f;
#pragma unroll
    for (int k = 0; k < 8; ++k) s += red[k];
    sinv = 1.0f / sqrtf(fmaxf(s / (float)D, 1e-5f) + 1e-5f);
  }
  __syncthreads();
  const float r = sinv;
  for (int i = tid; i < D; i += 256) {
    float v = fin0(clip1e4(p[i]) * r * w[i]);
    out[(size_t)blockIdx.x * D + i] = (f16)v;
  }
}

// ---------- out = x + rmsnorm(a, w), f32 out ----------
__global__ __launch_bounds__(256) void residual_rmsnorm_kernel(
    const float* __restrict__ x, const float* __restrict__ a,
    const float* __restrict__ w, float* __restrict__ out, int D) {
  __shared__ float red[8];
  __shared__ float sinv;
  const int tid = threadIdx.x;
  const float* p = a + (size_t)blockIdx.x * D;
  const float* px = x + (size_t)blockIdx.x * D;
  float ss = 0.f;
  for (int i = tid; i < D; i += 256) { float v = clip1e4(p[i]); ss += v * v; }
  ss = waveReduceAdd(ss);
  if ((tid & 31) == 0) red[tid >> 5] = ss;
  __syncthreads();
  if (tid == 0) {
    float s = 0.f;
#pragma unroll
    for (int k = 0; k < 8; ++k) s += red[k];
    sinv = 1.0f / sqrtf(fmaxf(s / (float)D, 1e-5f) + 1e-5f);
  }
  __syncthreads();
  const float r = sinv;
  for (int i = tid; i < D; i += 256) {
    float v = fin0(clip1e4(p[i]) * r * w[i]);
    out[(size_t)blockIdx.x * D + i] = px[i] + v;
  }
}

// ---------- per-head (64-elem) rmsnorm, f32 -> f16, one wave per head row ----------
__global__ __launch_bounds__(256) void head_rmsnorm_f16_kernel(
    const float* __restrict__ in, const float* __restrict__ w,
    f16* __restrict__ out, int rows) {
  const int row = blockIdx.x * 8 + (threadIdx.x >> 5);
  const int lane = threadIdx.x & 31;
  if (row >= rows) return;
  const float* p = in + (size_t)row * 64;
  const float a = clip1e4(p[lane]);
  const float b = clip1e4(p[lane + 32]);
  float ss = waveReduceAdd(a * a + b * b);
  const float rinv = 1.f / sqrtf(fmaxf(ss * (1.f / 64.f), 1e-5f) + 1e-5f);
  out[(size_t)row * 64 + lane]      = (f16)fin0(a * rinv * w[lane]);
  out[(size_t)row * 64 + lane + 32] = (f16)fin0(b * rinv * w[lane + 32]);
}

// ---------- g = silu(z1) * z3 ----------
__global__ __launch_bounds__(256) void silu_mul_kernel(
    const f16* __restrict__ z1, const f16* __restrict__ z3,
    f16* __restrict__ g, size_t n) {
  const size_t stride = (size_t)gridDim.x * blockDim.x;
  for (size_t i = (size_t)blockIdx.x * blockDim.x + threadIdx.x; i < n; i += stride) {
    const float a = (float)z1[i];
    const float s = a / (1.f + __expf(-a));
    g[i] = (f16)(s * (float)z3[i]);
  }
}

// ---------- generic WMMA GEMM (global-load fed; documented fallback) ----------
__global__ __launch_bounds__(256) void gemm_wmma_f16_kernel(
    const f16* __restrict__ A, const f16* __restrict__ W,
    const float* __restrict__ bias, int nbias,
    float* __restrict__ Cf, f16* __restrict__ Ch,
    int M, int N, int K) {
  const int lane = threadIdx.x & 31;
  const int wave = threadIdx.x >> 5;
  const int hi = lane >> 4;
  const int l15 = lane & 15;
  const int m0 = blockIdx.x * 32 + (wave & 1) * 16;
  const int n0 = blockIdx.y * 256 + (wave >> 1) * 64;

  v8f acc[4] = {};
  const f16* Ap = A + (size_t)(m0 + l15) * K;
  const f16* Wp[4];
#pragma unroll
  for (int t = 0; t < 4; ++t) Wp[t] = W + (size_t)(n0 + t * 16 + l15) * K;

  for (int k0 = 0; k0 < K; k0 += 32) {
    const int ka = k0 + hi * 16;
    v8h alo = *(const v8h*)(Ap + ka);
    v8h ahi = *(const v8h*)(Ap + ka + 8);
    v16h a;
#pragma unroll
    for (int i = 0; i < 8; ++i) { a[i] = alo[i]; a[8 + i] = ahi[i]; }
    const int kb = k0 + hi * 16;
#pragma unroll
    for (int t = 0; t < 4; ++t) {
      v16h b = *(const v16h*)(Wp[t] + kb);
      acc[t] = __builtin_amdgcn_wmma_f32_16x16x32_f16(
          false, a, false, b, (short)0, acc[t], false, false);
    }
  }
#pragma unroll
  for (int t = 0; t < 4; ++t) {
    const int n = n0 + t * 16 + l15;
    const float bv = (bias && n < nbias) ? bias[n] : 0.f;
#pragma unroll
    for (int r = 0; r < 8; ++r) {
      const int m = m0 + r + hi * 8;
      const float v = acc[t][r] + bv;
      if (Cf) Cf[(size_t)m * N + n] = v;
      if (Ch) Ch[(size_t)m * N + n] = (f16)v;
    }
  }
}

// ---------- TDM-staged WMMA GEMM: C[M,N] = A[M,K](f16) @ W[N,K](f16)^T + bias ----------
// Block tile 64M x 256N, 8 waves, wave tile 16M x 128N (8 accumulators).
// B-tiles (256 rows x 32 halves per K-step) are DMA'd to LDS by the Tensor Data
// Mover, double-buffered; waves only issue A loads + LDS B-fragment reads.
// Requires M%64==0, N%256==0, K%32==0.
__global__ __launch_bounds__(256) void gemm_wmma_tdm_kernel(
    const f16* __restrict__ A, const f16* __restrict__ W,
    const float* __restrict__ bias, int nbias,
    float* __restrict__ Cf, f16* __restrict__ Ch,
    int M, int N, int K) {
  __shared__ __align__(128) f16 btile[2][256 * BT_ROW];  // 2 x 20KB
  const int lane = threadIdx.x & 31;
  const int wave = threadIdx.x >> 5;
  const int hi = lane >> 4, l15 = lane & 15;
  const int m0 = blockIdx.x * 64 + (wave & 3) * 16;
  const int nblk = blockIdx.y * 256;
  const int n0w = (wave >> 2) * 128;  // N offset of this wave inside block tile

  v8f acc[8] = {};
  const f16* Ap = A + (size_t)(m0 + l15) * K;
  const f16* Wb = W + (size_t)nblk * K;
  const int nk = K / 32;

  if (wave == 0)
    tdm_load_tile_f16((unsigned)(size_t)&btile[0][0], Wb, 256, K);

  for (int k = 0; k < nk; ++k) {
    if (wave == 0) {
      if (k + 1 < nk) {
        tdm_load_tile_f16((unsigned)(size_t)&btile[(k + 1) & 1][0],
                          Wb + (size_t)(k + 1) * 32, 256, K);
        __builtin_amdgcn_s_wait_tensorcnt(1);  // in-order: tile k complete
      } else {
        __builtin_amdgcn_s_wait_tensorcnt(0);
      }
    }
    __syncthreads();  // tile k visible to all waves

    const int ka = k * 32 + hi * 16;
    __builtin_prefetch(Ap + ka + 64, 0, 0);  // global_prefetch_b8, 2 K-steps ahead
    v8h alo = *(const v8h*)(Ap + ka);
    v8h ahi = *(const v8h*)(Ap + ka + 8);
    v16h a;
#pragma unroll
    for (int i = 0; i < 8; ++i) { a[i] = alo[i]; a[8 + i] = ahi[i]; }

    const f16* bt = &btile[k & 1][0];
#pragma unroll
    for (int t = 0; t < 8; ++t) {
      const f16* bp = bt + (size_t)(n0w + t * 16 + l15) * BT_ROW + hi * 16;
      v8h blo = *(const v8h*)bp;
      v8h bhi = *(const v8h*)(bp + 8);
      v16h b;
#pragma unroll
      for (int i = 0; i < 8; ++i) { b[i] = blo[i]; b[8 + i] = bhi[i]; }
      acc[t] = __builtin_amdgcn_wmma_f32_16x16x32_f16(
          false, a, false, b, (short)0, acc[t], false, false);
    }
    __syncthreads();  // all waves done with tile k before its buffer is refilled
  }

#pragma unroll
  for (int t = 0; t < 8; ++t) {
    const int n = nblk + n0w + t * 16 + l15;
    const float bv = (bias && n < nbias) ? bias[n] : 0.f;
#pragma unroll
    for (int r = 0; r < 8; ++r) {
      const int m = m0 + r + hi * 8;  // C layout: VGPR r -> row r / r+8 per half-wave
      const float v = acc[t][r] + bv;
      if (Cf) Cf[(size_t)m * N + n] = v;
      if (Ch) Ch[(size_t)m * N + n] = (f16)v;
    }
  }
}

// ---------- flash attention, windowed causal, one wave per 16-query tile ----------
// Q,K,V,O layout: [B, L, H, 64] f16. grid = (L/64, H, B), block = 128 (4 waves).
__global__ __launch_bounds__(128) void attn_wmma_kernel(
    const f16* __restrict__ Q, const f16* __restrict__ Km,
    const f16* __restrict__ V, f16* __restrict__ O,
    const int* __restrict__ winp, int Ln, int Hn) {
  __shared__ f16 pbuf[4][16 * 32];
  const int win = *winp;
  const int lane = threadIdx.x & 31;
  const int wave = threadIdx.x >> 5;
  const int hi = lane >> 4, l15 = lane & 15;
  const int qbase = (blockIdx.x * 4 + wave) * 16;
  const int h = blockIdx.y, b = blockIdx.z;
  const size_t tstride = (size_t)Hn * 64;
  const f16* Qb = Q + ((size_t)b * Ln) * tstride + (size_t)h * 64;
  const f16* Kb = Km + ((size_t)b * Ln) * tstride + (size_t)h * 64;
  const f16* Vb = V + ((size_t)b * Ln) * tstride + (size_t)h * 64;
  f16* Ob = O + ((size_t)b * Ln) * tstride + (size_t)h * 64;

  // Q A-fragments for the two 32-wide K-dim steps of HD=64
  v16h qa[2];
  {
    const f16* qp = Qb + (size_t)(qbase + l15) * tstride;
#pragma unroll
    for (int s = 0; s < 2; ++s) {
      const int ka = s * 32 + hi * 16;
      v8h lo = *(const v8h*)(qp + ka);
      v8h hh = *(const v8h*)(qp + ka + 8);
#pragma unroll
      for (int i = 0; i < 8; ++i) { qa[s][i] = lo[i]; qa[s][8 + i] = hh[i]; }
    }
  }

  float mrun[8], lrun[8];
#pragma unroll
  for (int r = 0; r < 8; ++r) { mrun[r] = -1e30f; lrun[r] = 0.f; }
  v8f oacc[4] = {};
  int jlo = qbase + 15 - (win - 1);
  if (jlo < 0) jlo = 0;
  jlo &= ~31;
  const int jhi = qbase + 15;
  f16* pb = pbuf[wave];

  for (int j0 = jlo; j0 <= jhi; j0 += 32) {
    // S = Q @ K^T for 16 queries x 32 keys (two 16-wide N tiles)
    v8f s0 = {}, s1 = {};
#pragma unroll
    for (int s = 0; s < 2; ++s) {
      const int kb = s * 32 + hi * 16;
      v16h b0 = *(const v16h*)(Kb + (size_t)(j0 + l15) * tstride + kb);
      v16h b1 = *(const v16h*)(Kb + (size_t)(j0 + 16 + l15) * tstride + kb);
      s0 = __builtin_amdgcn_wmma_f32_16x16x32_f16(false, qa[s], false, b0, (short)0, s0, false, false);
      s1 = __builtin_amdgcn_wmma_f32_16x16x32_f16(false, qa[s], false, b1, (short)0, s1, false, false);
    }
    const float scale = 0.125f;  // 64^-0.5
#pragma unroll
    for (int r = 0; r < 8; ++r) {
      const int qi = qbase + r + hi * 8;
      const int k0i = j0 + l15, k1i = k0i + 16;
      const bool bad0 = (k0i > qi) || (qi - k0i >= win);
      const bool bad1 = (k1i > qi) || (qi - k1i >= win);
      s0[r] = bad0 ? -1e30f : s0[r] * scale;
      s1[r] = bad1 ? -1e30f : s1[r] * scale;
      const float mt = halfReduceMax(fmaxf(s0[r], s1[r]));
      const float mnew = fmaxf(mrun[r], mt);
      const float corr = __expf(mrun[r] - mnew);
      const float p0 = (s0[r] <= -1e29f) ? 0.f : __expf(s0[r] - mnew);
      const float p1 = (s1[r] <= -1e29f) ? 0.f : __expf(s1[r] - mnew);
      const float rs = halfReduceAdd(p0 + p1);
      lrun[r] = lrun[r] * corr + rs;
      mrun[r] = mnew;
#pragma unroll
      for (int t = 0; t < 4; ++t) oacc[t][r] *= corr;
      const int rowl = r + hi * 8;
      pb[rowl * 32 + l15] = (f16)p0;       // transpose C-layout -> LDS row-major
      pb[rowl * 32 + 16 + l15] = (f16)p1;
    }
    asm volatile("s_wait_dscnt 0" ::: "memory");  // LDS stores visible before A-frag reload
    // P as A-fragment (16 queries x 32 keys)
    v16h pa;
    {
      const int ka = hi * 16;
      const f16* pr = pb + l15 * 32;
      v8h lo = *(const v8h*)(pr + ka);
      v8h hh = *(const v8h*)(pr + ka + 8);
#pragma unroll
      for (int i = 0; i < 8; ++i) { pa[i] = lo[i]; pa[8 + i] = hh[i]; }
    }
    // V B-fragments: B[k][n] = V[j0+k][dim n] -> per-lane strided gather
    const int kb = hi * 16;
#pragma unroll
    for (int t = 0; t < 4; ++t) {
      v16h bv;
      const f16* vp = Vb + (size_t)(j0 + kb) * tstride + t * 16 + l15;
#pragma unroll
      for (int j = 0; j < 16; ++j) bv[j] = vp[(size_t)j * tstride];
      oacc[t] = __builtin_amdgcn_wmma_f32_16x16x32_f16(false, pa, false, bv, (short)0, oacc[t], false, false);
    }
  }
#pragma unroll
  for (int r = 0; r < 8; ++r) {
    const int rowl = r + hi * 8;
    const float linv = (lrun[r] > 0.f) ? (1.f / lrun[r]) : 0.f;
#pragma unroll
    for (int t = 0; t < 4; ++t) {
      const float v = fin0(oacc[t][r] * linv);
      Ob[(size_t)(qbase + rowl) * tstride + t * 16 + l15] = (f16)v;
    }
  }
}

// ---------- launcher ----------
extern "C" void kernel_launch(void* const* d_in, const int* in_sizes, int n_in,
                              void* d_out, int out_size, void* d_ws, size_t ws_size,
                              hipStream_t stream) {
  (void)in_sizes; (void)n_in; (void)out_size; (void)ws_size;
  constexpr int Bc = 4, Lc = 1024, Dc = 1024, Hc = 16;
  constexpr int Mc = Bc * Lc;          // 4096 token rows
  constexpr int Ic = 3280, IP = 3328;  // padded intermediate (mult of 256)

  const float* x        = (const float*)d_in[0];
  const float* wq_w     = (const float*)d_in[1];
  const float* wq_b     = (const float*)d_in[2];
  const float* wk_w     = (const float*)d_in[3];
  const float* wk_b     = (const float*)d_in[4];
  const float* wv_w     = (const float*)d_in[5];
  const float* wv_b     = (const float*)d_in[6];
  const float* wo_w     = (const float*)d_in[7];
  const float* wo_b     = (const float*)d_in[8];
  const float* q_norm_w = (const float*)d_in[9];
  const float* k_norm_w = (const float*)d_in[10];
  const float* seq_norm_w      = (const float*)d_in[11];
  const float* seq_post_norm_w = (const float*)d_in[12];
  const float* ffn_norm_w      = (const float*)d_in[13];
  const float* ffn_post_norm_w = (const float*)d_in[14];
  const float* w1_w = (const float*)d_in[15];
  const float* w1_b = (const float*)d_in[16];
  const float* w2_w = (const float*)d_in[17];
  const float* w2_b = (const float*)d_in[18];
  const float* w3_w = (const float*)d_in[19];
  const float* w3_b = (const float*)d_in[20];
  const int*   winp = (const int*)d_in[21];
  float* out = (float*)d_out;

  char* ws = (char*)d_ws;
  size_t off = 0;
  auto alloc = [&](size_t bytes) -> void* {
    void* p = ws + off;
    off = (off + bytes + 255) & ~(size_t)255;
    return p;
  };
  f16* wq16 = (f16*)alloc((size_t)Dc * Dc * 2);
  f16* wk16 = (f16*)alloc((size_t)Dc * Dc * 2);
  f16* wv16 = (f16*)alloc((size_t)Dc * Dc * 2);
  f16* wo16 = (f16*)alloc((size_t)Dc * Dc * 2);
  f16* w116 = (f16*)alloc((size_t)IP * Dc * 2);
  f16* w316 = (f16*)alloc((size_t)IP * Dc * 2);
  f16* w216 = (f16*)alloc((size_t)Dc * IP * 2);
  f16* xn16 = (f16*)alloc((size_t)Mc * Dc * 2);
  float* qf = (float*)alloc((size_t)Mc * Dc * 4);
  float* kf = (float*)alloc((size_t)Mc * Dc * 4);
  f16* q16 = (f16*)alloc((size_t)Mc * Dc * 2);
  f16* k16 = (f16*)alloc((size_t)Mc * Dc * 2);
  f16* v16buf = (f16*)alloc((size_t)Mc * Dc * 2);
  f16* attn16 = (f16*)alloc((size_t)Mc * Dc * 2);
  float* aout = (float*)alloc((size_t)Mc * Dc * 4);
  float* hbuf = (float*)alloc((size_t)Mc * Dc * 4);
  f16* hn16 = (f16*)alloc((size_t)Mc * Dc * 2);
  f16* z116 = (f16*)alloc((size_t)Mc * IP * 2);
  f16* z316 = (f16*)alloc((size_t)Mc * IP * 2);
  f16* g16  = (f16*)alloc((size_t)Mc * IP * 2);
  float* fbuf = (float*)alloc((size_t)Mc * Dc * 4);

  // weight conversion / padding (L2-resident f16 copies)
  cvt_pad_f16_kernel<<<2048, 256, 0, stream>>>(wq_w, wq16, Dc, Dc, Dc, Dc);
  cvt_pad_f16_kernel<<<2048, 256, 0, stream>>>(wk_w, wk16, Dc, Dc, Dc, Dc);
  cvt_pad_f16_kernel<<<2048, 256, 0, stream>>>(wv_w, wv16, Dc, Dc, Dc, Dc);
  cvt_pad_f16_kernel<<<2048, 256, 0, stream>>>(wo_w, wo16, Dc, Dc, Dc, Dc);
  cvt_pad_f16_kernel<<<4096, 256, 0, stream>>>(w1_w, w116, Ic, Dc, IP, Dc);
  cvt_pad_f16_kernel<<<4096, 256, 0, stream>>>(w3_w, w316, Ic, Dc, IP, Dc);
  cvt_pad_f16_kernel<<<4096, 256, 0, stream>>>(w2_w, w216, Dc, Ic, Dc, IP);

  // xn = rmsnorm(x, seq_norm_w)
  rmsnorm_row_f16_kernel<<<Mc, 256, 0, stream>>>(x, seq_norm_w, xn16, Dc);

  // q,k,v projections (TDM-staged WMMA GEMM)
  dim3 gDD(Mc / 64, Dc / 256);
  gemm_wmma_tdm_kernel<<<gDD, 256, 0, stream>>>(xn16, wq16, wq_b, Dc, qf, nullptr, Mc, Dc, Dc);
  gemm_wmma_tdm_kernel<<<gDD, 256, 0, stream>>>(xn16, wk16, wk_b, Dc, kf, nullptr, Mc, Dc, Dc);
  gemm_wmma_tdm_kernel<<<gDD, 256, 0, stream>>>(xn16, wv16, wv_b, Dc, nullptr, v16buf, Mc, Dc, Dc);

  // per-head rmsnorm -> f16 q,k
  head_rmsnorm_f16_kernel<<<(Mc * Hc) / 8, 256, 0, stream>>>(qf, q_norm_w, q16, Mc * Hc);
  head_rmsnorm_f16_kernel<<<(Mc * Hc) / 8, 256, 0, stream>>>(kf, k_norm_w, k16, Mc * Hc);

  // windowed flash attention
  attn_wmma_kernel<<<dim3(Lc / 64, Hc, Bc), 128, 0, stream>>>(q16, k16, v16buf, attn16, winp, Lc, Hc);

  // output projection + post-norm residual
  gemm_wmma_tdm_kernel<<<gDD, 256, 0, stream>>>(attn16, wo16, wo_b, Dc, aout, nullptr, Mc, Dc, Dc);
  residual_rmsnorm_kernel<<<Mc, 256, 0, stream>>>(x, aout, seq_post_norm_w, hbuf, Dc);

  // FFN
  rmsnorm_row_f16_kernel<<<Mc, 256, 0, stream>>>(hbuf, ffn_norm_w, hn16, Dc);
  dim3 gDI(Mc / 64, IP / 256);
  gemm_wmma_tdm_kernel<<<gDI, 256, 0, stream>>>(hn16, w116, w1_b, Ic, nullptr, z116, Mc, IP, Dc);
  gemm_wmma_tdm_kernel<<<gDI, 256, 0, stream>>>(hn16, w316, w3_b, Ic, nullptr, z316, Mc, IP, Dc);
  silu_mul_kernel<<<4096, 256, 0, stream>>>(z116, z316, g16, (size_t)Mc * IP);
  gemm_wmma_tdm_kernel<<<gDD, 256, 0, stream>>>(g16, w216, w2_b, Dc, fbuf, nullptr, Mc, Dc, IP);

  // final residual
  residual_rmsnorm_kernel<<<Mc, 256, 0, stream>>>(hbuf, fbuf, ffn_post_norm_w, out, Dc);
}
